// MHAttention_31086973288604
// MI455X (gfx1250) — matrix operand order repaired and checked by reading, not compiled, not run on hardware
//
#include <hip/hip_runtime.h>
#include <hip/hip_bf16.h>

#define DIM    1024
#define NHEADS 16
#define HDIM   64
#define BATCH  4
#define SEQ    1024
#define BT     (BATCH * SEQ)   // 4096 rows

typedef __bf16 bf16;
typedef __attribute__((ext_vector_type(16))) __bf16 v16bf;
typedef __attribute__((ext_vector_type(8)))  __bf16 v8bf;
typedef __attribute__((ext_vector_type(8)))  float  v8f;
typedef __attribute__((ext_vector_type(4)))  float  v4f;

__device__ __forceinline__ bf16 f2bf(float f) { return (bf16)f; }

__device__ __forceinline__ v8f wmma_bf16(v16bf a, v16bf b, v8f c) {
  // D = A(16x32) * B(32x16) + C, fp32 accumulate
  return __builtin_amdgcn_wmma_f32_16x16x32_bf16(
      /*neg_a=*/false, a, /*neg_b=*/false, b,
      /*c_mod=*/(short)0, c, /*reuse_a=*/false, /*reuse_b=*/false);
}

// ---- Fragment loaders (contiguous k order, same for A and B), bf16 sources only ----------
// A 16x32: row M = lane&15; elem i -> K = half*16 + i (contiguous)
__device__ __forceinline__ v16bf loadA_bf16(const bf16* __restrict__ src, int ld,
                                            int m, int half, int k0) {
  v16bf a;
  const bf16* p = src + (size_t)m * ld + k0;
#pragma unroll
  for (int i = 0; i < 8; ++i) a[i] = p[half * 16 + i];
#pragma unroll
  for (int i = 0; i < 8; ++i) a[8 + i] = p[half * 16 + 8 + i];
  return a;
}
// B 32x16: col N = lane&15 = source row `n`; elem i -> K = half*16 + i (contiguous)
__device__ __forceinline__ v16bf loadB_bf16(const bf16* __restrict__ src, int ld,
                                            int n, int half, int k0) {
  v16bf b;
  const bf16* p = src + (size_t)n * ld + k0 + half * 16;
#pragma unroll
  for (int i = 0; i < 16; ++i) b[i] = p[i];
  return b;
}

__device__ __forceinline__ v8f vzero8() {
  v8f z = {0.f, 0.f, 0.f, 0.f, 0.f, 0.f, 0.f, 0.f};
  return z;
}

// ================= Kernel 0: one-shot fp32 -> bf16 conversion ============
// Reads each fp32 exactly once; all later GEMM passes stream bf16 from L2.
__global__ __launch_bounds__(256) void cvt_bf16_kernel(
    const float* __restrict__ src, bf16* __restrict__ dst, int n) {
  const int i = (blockIdx.x * 256 + threadIdx.x) * 8;
  if (i >= n) return;
  v4f a = *(const v4f*)(src + i);
  v4f b = *(const v4f*)(src + i + 4);
  v8bf o;
#pragma unroll
  for (int j = 0; j < 4; ++j) o[j] = f2bf(a[j]);
#pragma unroll
  for (int j = 0; j < 4; ++j) o[4 + j] = f2bf(b[j]);
  *(v8bf*)(dst + i) = o;
}

// ======================= Kernel 1: QKV projection ========================
// C[m, n] = sum_k X[m,k] * W_qkv[n,k] + b_qkv[n] ; m in [0,4096), n in [0,3072)
// One wave computes a 32x64 tile (2x4 WMMA accs; 8 WMMAs per 6 frag loads).
// Scatter epilogue: n<1024 -> Q[b,h,t,d]; n<2048 -> K[b,h,t,d]; else V[b,h,d,t].
__global__ __launch_bounds__(32) void qkv_proj_kernel(
    const bf16* __restrict__ X, const bf16* __restrict__ W,
    const float* __restrict__ bias,
    bf16* __restrict__ qbuf, bf16* __restrict__ kbuf, bf16* __restrict__ vbuf) {
  const int lane = threadIdx.x & 31;
  const int half = lane >> 4;
  const int ln   = lane & 15;
  const int m0 = blockIdx.x * 32;
  const int n0 = blockIdx.y * 64;

  v8f acc[2][4];
#pragma unroll
  for (int i = 0; i < 2; ++i)
#pragma unroll
    for (int j = 0; j < 4; ++j) acc[i][j] = vzero8();

  for (int k0 = 0; k0 < DIM; k0 += 32) {
    v16bf a0 = loadA_bf16(X, DIM, m0 + ln,      half, k0);
    v16bf a1 = loadA_bf16(X, DIM, m0 + 16 + ln, half, k0);
    v16bf b[4];
#pragma unroll
    for (int ni = 0; ni < 4; ++ni)
      b[ni] = loadB_bf16(W, DIM, n0 + ni * 16 + ln, half, k0);
#pragma unroll
    for (int ni = 0; ni < 4; ++ni) {
      acc[0][ni] = wmma_bf16(a0, b[ni], acc[0][ni]);
      acc[1][ni] = wmma_bf16(a1, b[ni], acc[1][ni]);
    }
  }

#pragma unroll
  for (int mi = 0; mi < 2; ++mi)
#pragma unroll
    for (int ni = 0; ni < 4; ++ni)
#pragma unroll
      for (int r = 0; r < 8; ++r) {
        const int m = m0 + mi * 16 + r + 8 * half;   // global row over B*T
        const int n = n0 + ni * 16 + ln;             // global col over 3*DIM
        const float val = acc[mi][ni][r] + bias[n];
        const int b = m >> 10, t = m & (SEQ - 1);
        const int which = n >> 10;                   // 0:Q 1:K 2:V
        const int nn = n & (DIM - 1);
        const int h = nn >> 6, d = nn & (HDIM - 1);
        const bf16 bv = f2bf(val);
        const size_t bh = (size_t)b * NHEADS + h;
        if (which == 0)      qbuf[(bh * SEQ + t) * HDIM + d] = bv;
        else if (which == 1) kbuf[(bh * SEQ + t) * HDIM + d] = bv;
        else                 vbuf[(bh * HDIM + d) * SEQ + t] = bv;  // V^T
      }
}

// ======================= Kernel 2: flash attention =======================
// One wave handles 32 query rows (2 q-tiles) of one (b,h); streams keys in
// blocks of 32. K/V fragments are loaded once and reused by both q-tiles.
// S = (Q K^T)/8 via WMMA; online softmax; P re-swizzled C->A via LDS;
// O += P V via WMMA (hd = 64 -> 4 N-tiles).
__global__ __launch_bounds__(32) void attn_kernel(
    const bf16* __restrict__ qbuf, const bf16* __restrict__ kbuf,
    const bf16* __restrict__ vbuf, bf16* __restrict__ aout) {
  __shared__ bf16 Pbuf[2][16 * 32];  // 2 KB: P tiles (C-layout -> A-layout)

  const int lane = threadIdx.x & 31;
  const int half = lane >> 4;
  const int ln   = lane & 15;
  const int q0 = blockIdx.x * 32;
  const int bh = blockIdx.y;  // b*NHEADS + h

  const bf16* Qh = qbuf + (size_t)bh * SEQ * HDIM;
  const bf16* Kh = kbuf + (size_t)bh * SEQ * HDIM;
  const bf16* Vt = vbuf + (size_t)bh * HDIM * SEQ;  // [64][1024]

  // Q fragments persist for the whole key loop (hd=64 -> 2 K-chunks of 32)
  v16bf qf[2][2];
#pragma unroll
  for (int qt = 0; qt < 2; ++qt) {
    qf[qt][0] = loadA_bf16(Qh, HDIM, q0 + qt * 16 + ln, half, 0);
    qf[qt][1] = loadA_bf16(Qh, HDIM, q0 + qt * 16 + ln, half, 32);
  }

  v8f oacc[2][4];
#pragma unroll
  for (int qt = 0; qt < 2; ++qt)
#pragma unroll
    for (int dt = 0; dt < 4; ++dt) oacc[qt][dt] = vzero8();
  float rowmax[2][8], rowsum[2][8];
#pragma unroll
  for (int qt = 0; qt < 2; ++qt)
#pragma unroll
    for (int r = 0; r < 8; ++r) { rowmax[qt][r] = -3.0e38f; rowsum[qt][r] = 0.f; }

  for (int kblk = 0; kblk < SEQ; kblk += 32) {
    if (kblk + 32 < SEQ) {  // global_prefetch_b8 next key/value block
      __builtin_prefetch(Kh + (size_t)(kblk + 32 + lane) * HDIM, 0, 0);
      __builtin_prefetch(Vt + (size_t)(lane << 1) * SEQ + kblk + 32, 0, 0);
    }
    // ---- scores: two 16x16 sub-tiles x two q-tiles ----
    v8f S[2][2];
#pragma unroll
    for (int s = 0; s < 2; ++s) {
      v16bf kf0 = loadB_bf16(Kh, HDIM, kblk + s * 16 + ln, half, 0);
      v16bf kf1 = loadB_bf16(Kh, HDIM, kblk + s * 16 + ln, half, 32);
#pragma unroll
      for (int qt = 0; qt < 2; ++qt) {
        v8f t = vzero8();
        t = wmma_bf16(qf[qt][0], kf0, t);
        t = wmma_bf16(qf[qt][1], kf1, t);
#pragma unroll
        for (int r = 0; r < 8; ++r) t[r] *= 0.125f;  // 1/sqrt(64)
        S[qt][s] = t;
      }
    }
    // ---- online softmax per q-tile ----
#pragma unroll
    for (int qt = 0; qt < 2; ++qt) {
      float bm[8];
#pragma unroll
      for (int r = 0; r < 8; ++r) bm[r] = fmaxf(S[qt][0][r], S[qt][1][r]);
      for (int off = 1; off < 16; off <<= 1)
#pragma unroll
        for (int r = 0; r < 8; ++r) bm[r] = fmaxf(bm[r], __shfl_xor(bm[r], off, 32));
      float scale[8];
#pragma unroll
      for (int r = 0; r < 8; ++r) {
        const float nm = fmaxf(rowmax[qt][r], bm[r]);
        scale[r] = __expf(rowmax[qt][r] - nm);
        rowmax[qt][r] = nm;
      }
#pragma unroll
      for (int dt = 0; dt < 4; ++dt)
#pragma unroll
        for (int r = 0; r < 8; ++r) oacc[qt][dt][r] *= scale[r];
      float bs[8];
#pragma unroll
      for (int r = 0; r < 8; ++r) {
        const float p0 = __expf(S[qt][0][r] - rowmax[qt][r]);
        const float p1 = __expf(S[qt][1][r] - rowmax[qt][r]);
        const int m = r + 8 * half;
        Pbuf[qt][m * 32 + ln]      = f2bf(p0);
        Pbuf[qt][m * 32 + 16 + ln] = f2bf(p1);
        bs[r] = p0 + p1;
      }
      for (int off = 1; off < 16; off <<= 1)
#pragma unroll
        for (int r = 0; r < 8; ++r) bs[r] += __shfl_xor(bs[r], off, 32);
#pragma unroll
      for (int r = 0; r < 8; ++r) rowsum[qt][r] = rowsum[qt][r] * scale[r] + bs[r];
    }
    __syncthreads();
    // ---- O += P(16x32) * V(32x64); V fragments shared across q-tiles ----
    v16bf pf[2];
#pragma unroll
    for (int qt = 0; qt < 2; ++qt) pf[qt] = loadA_bf16(Pbuf[qt], 32, ln, half, 0);
#pragma unroll
    for (int dt = 0; dt < 4; ++dt) {
      v16bf vf = loadB_bf16(Vt, SEQ, dt * 16 + ln, half, kblk);
      oacc[0][dt] = wmma_bf16(pf[0], vf, oacc[0][dt]);
      oacc[1][dt] = wmma_bf16(pf[1], vf, oacc[1][dt]);
    }
    __syncthreads();
  }

  // normalize + store in [B, T, H, hd] (== [B*T, DIM] row-major for out-proj)
  const int b = bh / NHEADS, h = bh % NHEADS;
#pragma unroll
  for (int qt = 0; qt < 2; ++qt)
#pragma unroll
    for (int dt = 0; dt < 4; ++dt)
#pragma unroll
      for (int r = 0; r < 8; ++r) {
        const int m = r + 8 * half;
        const int t = q0 + qt * 16 + m;
        const int d = dt * 16 + ln;
        const float val = oacc[qt][dt][r] / rowsum[qt][r];
        aout[((size_t)(b * SEQ + t) * NHEADS + h) * HDIM + d] = f2bf(val);
      }
}

// ======================= Kernel 3: output projection =====================
// out[m, n] = sum_k A[m,k] * W_out[n,k] + b_out[n]  (fp32 result)
// One wave computes a 32x64 tile.
__global__ __launch_bounds__(32) void out_proj_kernel(
    const bf16* __restrict__ Ain, const bf16* __restrict__ W,
    const float* __restrict__ bias, float* __restrict__ out) {
  const int lane = threadIdx.x & 31;
  const int half = lane >> 4;
  const int ln   = lane & 15;
  const int m0 = blockIdx.x * 32;
  const int n0 = blockIdx.y * 64;

  v8f acc[2][4];
#pragma unroll
  for (int i = 0; i < 2; ++i)
#pragma unroll
    for (int j = 0; j < 4; ++j) acc[i][j] = vzero8();

  for (int k0 = 0; k0 < DIM; k0 += 32) {
    v16bf a0 = loadA_bf16(Ain, DIM, m0 + ln,      half, k0);
    v16bf a1 = loadA_bf16(Ain, DIM, m0 + 16 + ln, half, k0);
    v16bf b[4];
#pragma unroll
    for (int ni = 0; ni < 4; ++ni)
      b[ni] = loadB_bf16(W, DIM, n0 + ni * 16 + ln, half, k0);
#pragma unroll
    for (int ni = 0; ni < 4; ++ni) {
      acc[0][ni] = wmma_bf16(a0, b[ni], acc[0][ni]);
      acc[1][ni] = wmma_bf16(a1, b[ni], acc[1][ni]);
    }
  }

#pragma unroll
  for (int mi = 0; mi < 2; ++mi)
#pragma unroll
    for (int ni = 0; ni < 4; ++ni)
#pragma unroll
      for (int r = 0; r < 8; ++r) {
        const int m = m0 + mi * 16 + r + 8 * half;
        const int n = n0 + ni * 16 + ln;
        out[(size_t)m * DIM + n] = acc[mi][ni][r] + bias[n];
      }
}

// ================================ launch =================================
extern "C" void kernel_launch(void* const* d_in, const int* in_sizes, int n_in,
                              void* d_out, int out_size, void* d_ws, size_t ws_size,
                              hipStream_t stream) {
  (void)in_sizes; (void)n_in; (void)out_size; (void)ws_size;
  const float* x     = (const float*)d_in[0];
  const float* W_qkv = (const float*)d_in[1];
  const float* b_qkv = (const float*)d_in[2];
  const float* W_out = (const float*)d_in[3];
  const float* b_out = (const float*)d_in[4];
  float* out = (float*)d_out;

  // workspace layout (bf16 elements):
  //   qbuf/kbuf/vbuf/abuf : 4 x B*H*T*hd = 4 x 4M             (32 MB)
  //   Xb   : BT*DIM   = 4M   Wqkvb : 3*DIM*DIM = 3M   Woutb : DIM*DIM = 1M
  const size_t per = (size_t)BATCH * NHEADS * SEQ * HDIM;
  bf16* qbuf  = (bf16*)d_ws;
  bf16* kbuf  = qbuf  + per;
  bf16* vbuf  = kbuf  + per;          // stored transposed [B,H,hd,T]
  bf16* abuf  = vbuf  + per;          // attention output [B,T,H,hd]
  bf16* Xb    = abuf  + per;
  bf16* Wqkvb = Xb    + (size_t)BT * DIM;
  bf16* Woutb = Wqkvb + (size_t)3 * DIM * DIM;

  const int nX = BT * DIM, nWq = 3 * DIM * DIM, nWo = DIM * DIM;
  cvt_bf16_kernel<<<dim3((nX  + 2047) / 2048), 256, 0, stream>>>(x,     Xb,    nX);
  cvt_bf16_kernel<<<dim3((nWq + 2047) / 2048), 256, 0, stream>>>(W_qkv, Wqkvb, nWq);
  cvt_bf16_kernel<<<dim3((nWo + 2047) / 2048), 256, 0, stream>>>(W_out, Woutb, nWo);

  qkv_proj_kernel<<<dim3(BT / 32, 3 * DIM / 64), 32, 0, stream>>>(
      Xb, Wqkvb, b_qkv, qbuf, kbuf, vbuf);
  attn_kernel<<<dim3(SEQ / 32, BATCH * NHEADS), 32, 0, stream>>>(
      qbuf, kbuf, vbuf, abuf);
  out_proj_kernel<<<dim3(BT / 32, DIM / 64), 32, 0, stream>>>(
      abuf, Woutb, b_out, out);
}
